// VanillaLSTM_17789754540430
// MI455X (gfx1250) — matrix operand order repaired and checked by reading, not compiled, not run on hardware
//
#include <hip/hip_runtime.h>
#include <hip/hip_bf16.h>
#include <stdint.h>

// Problem constants (VanillaLSTM): B=64, S=512, I=512, H=1024, gates 4H=4096
#define BB 64
#define SS 512
#define II 512
#define HH 1024

typedef __attribute__((ext_vector_type(16))) __bf16 v16bf;
typedef __attribute__((ext_vector_type(8)))  float  v8f;

// ---------- helpers ----------
static __device__ __forceinline__ unsigned short f2bf_bits(float f) {
    union { float f; uint32_t u; } v; v.f = f;
    uint32_t u = v.u;
    uint32_t r = u + 0x7FFFu + ((u >> 16) & 1u);   // RNE
    return (unsigned short)(r >> 16);
}

// Build a v16bf fragment from two contiguous 16-byte chunks (8 bf16 each).
static __device__ __forceinline__ v16bf load_frag(const unsigned short* p0,
                                                  const unsigned short* p1) {
    uint4 a = *reinterpret_cast<const uint4*>(p0);
    uint4 b = *reinterpret_cast<const uint4*>(p1);
    uint32_t tmp[8] = {a.x, a.y, a.z, a.w, b.x, b.y, b.z, b.w};
    v16bf r;
    __builtin_memcpy(&r, tmp, sizeof(r));
    return r;
}

static __device__ __forceinline__ float sigmoid_f(float x) {
    return 1.0f / (1.0f + __expf(-x));
}
static __device__ __forceinline__ float tanh_f(float x) {
    x = fminf(15.0f, fmaxf(-15.0f, x));
    float e = __expf(2.0f * x);
    return (e - 1.0f) / (e + 1.0f);
}

// Software-pipelined K-slice GEMM: NK k-steps of 32, 4 gate accumulators.
//   arow : per-lane A pointer (row base + kstart*32 + aoff), runs in 32-elem steps
//   b0   : per-lane packed-B pointer for gate 0 at kstart (16 contiguous bf16/lane)
//   gstride : elements between consecutive gate blocks in packed W
template <int NK>
static __device__ __forceinline__ void gemm_accum(
    const unsigned short* __restrict__ arow,
    const unsigned short* __restrict__ b0,
    size_t gstride, v8f acc[4]) {
    v16bf a_cur = load_frag(arow, arow + 8);
    v16bf b_cur[4];
#pragma unroll
    for (int g = 0; g < 4; g++) {
        const unsigned short* bp = b0 + (size_t)g * gstride;
        b_cur[g] = load_frag(bp, bp + 8);
    }
#pragma unroll
    for (int ks = 1; ks < NK; ks++) {
        // issue next-fragment loads (independent regs) ...
        const unsigned short* ap = arow + ks * 32;
        v16bf a_nxt = load_frag(ap, ap + 8);
        v16bf b_nxt[4];
#pragma unroll
        for (int g = 0; g < 4; g++) {
            const unsigned short* bp = b0 + (size_t)g * gstride + (size_t)ks * 512;
            b_nxt[g] = load_frag(bp, bp + 8);
        }
        // ... while the 4 WMMAs for the current fragment execute
#pragma unroll
        for (int g = 0; g < 4; g++)
            acc[g] = __builtin_amdgcn_wmma_f32_16x16x32_bf16(
                false, a_cur, false, b_cur[g], (short)0, acc[g], false, false);
        a_cur = a_nxt;
#pragma unroll
        for (int g = 0; g < 4; g++) b_cur[g] = b_nxt[g];
    }
#pragma unroll
    for (int g = 0; g < 4; g++)
        acc[g] = __builtin_amdgcn_wmma_f32_16x16x32_bf16(
            false, a_cur, false, b_cur[g], (short)0, acc[g], false, false);
}

// ---------- conversion kernels ----------
__global__ void convert_x_kernel(const float* __restrict__ x,
                                 unsigned short* __restrict__ xb, int n4) {
    int id = blockIdx.x * blockDim.x + threadIdx.x;
    if (id >= n4) return;
    float4 v = reinterpret_cast<const float4*>(x)[id];
    union { unsigned short s[4]; uint2 u; } o;
    o.s[0] = f2bf_bits(v.x);
    o.s[1] = f2bf_bits(v.y);
    o.s[2] = f2bf_bits(v.z);
    o.s[3] = f2bf_bits(v.w);
    reinterpret_cast<uint2*>(xb)[id] = o.u;
}

// Pre-pack W [K,4096] (fp32, row-major) into WMMA B-fragment order (bf16):
// packed[(((ntile*nk)+ks)*32 + lane)*16 + e] = W[ks*32 + (lane<16?0:16) + e][ntile*16 + (lane&15)]
__global__ void pack_w_kernel(const float* __restrict__ W,
                              unsigned short* __restrict__ Wp, int nk) {
    int id = blockIdx.x * blockDim.x + threadIdx.x;
    int total = 256 * nk * 32;
    if (id >= total) return;
    int lane  = id & 31;
    int rest  = id >> 5;
    int ks    = rest % nk;
    int ntile = rest / nk;
    int c     = ntile * 16 + (lane & 15);
    int kbase = ks * 32 + ((lane < 16) ? 0 : 16);
    unsigned short* dst = Wp + (size_t)id * 16;
#pragma unroll
    for (int e = 0; e < 16; e++)
        dst[e] = f2bf_bits(W[(size_t)(kbase + e) * (4 * HH) + c]);
}

__global__ void init_state_kernel(unsigned short* __restrict__ h_bf,
                                  float* __restrict__ c_st) {
    int id = blockIdx.x * blockDim.x + threadIdx.x;
    if (id < BB * HH) {
        c_st[id] = 0.0f;
        h_bf[id] = 0;                 // buffer 0
        h_bf[id + BB * HH] = 0;       // buffer 1
    }
}

// ---------- fused LSTM step ----------
// Grid: 256 blocks (one 16x16 batch x hidden tile each) x 128 threads (4 waves).
// The 4 waves split the K=1536 reduction (x: 4 k-steps each, h: 8 k-steps each),
// reduce partials through LDS, then each wave cell-updates 2 of the 8 rows.
__global__ __launch_bounds__(128)
void lstm_step_kernel(const unsigned short* __restrict__ x_bf,
                      const unsigned short* __restrict__ wx_p,
                      const unsigned short* __restrict__ wh_p,
                      const float* __restrict__ bias,
                      const unsigned short* __restrict__ h_in,
                      unsigned short* __restrict__ h_out,
                      float* __restrict__ c_st,
                      float* __restrict__ out,
                      int t) {
    __shared__ float red[4][32][33];          // +1 pad: kills LDS bank conflicts

    const int lane = threadIdx.x & 31;
    const int wave = threadIdx.x >> 5;        // K-slice id 0..3
    const int tile = blockIdx.x;              // 0..255
    const int mt = tile >> 6;                 // batch tile 0..3
    const int nt = tile & 63;                 // hidden-col tile 0..63
    const int m0 = mt * 16;
    const int row = lane & 15;
    const bool hi = lane >= 16;
    const int aoff = hi ? 16 : 0;

    v8f acc[4];
#pragma unroll
    for (int g = 0; g < 4; g++)
#pragma unroll
        for (int i = 0; i < 8; i++) acc[g][i] = 0.0f;

    // ---- x_t @ W_x slice: k-steps [4*wave, 4*wave+4) of 16 ----
    {
        const unsigned short* ax =
            x_bf + ((size_t)(m0 + row) * SS + t) * II + (4 * wave) * 32 + aoff;
        const unsigned short* bx =
            wx_p + ((size_t)nt * (II / 32) * 32 + lane) * 16 + (size_t)(4 * wave) * 512;
        gemm_accum<4>(ax, bx, (size_t)64 * (II / 32) * 512, acc);
    }
    // ---- h_{t-1} @ W_h slice: k-steps [8*wave, 8*wave+8) of 32 ----
    {
        const unsigned short* ah =
            h_in + (size_t)(m0 + row) * HH + (8 * wave) * 32 + aoff;
        const unsigned short* bh =
            wh_p + ((size_t)nt * (HH / 32) * 32 + lane) * 16 + (size_t)(8 * wave) * 512;
        gemm_accum<8>(ah, bh, (size_t)64 * (HH / 32) * 512, acc);
    }

    // ---- cross-wave K reduction through LDS ----
#pragma unroll
    for (int g = 0; g < 4; g++)
#pragma unroll
        for (int r = 0; r < 8; r++)
            red[wave][lane][g * 8 + r] = acc[g][r];
    __syncthreads();

    // ---- fused cell update: wave handles rows r in {2*wave, 2*wave+1} ----
    const int gn = nt * 16 + (lane & 15);
    const float bvals[4] = {bias[gn], bias[HH + gn], bias[2 * HH + gn], bias[3 * HH + gn]};
    const int rbase = hi ? 8 : 0;
#pragma unroll
    for (int rr = 0; rr < 2; rr++) {
        const int r = wave * 2 + rr;
        float gsum[4];
#pragma unroll
        for (int g = 0; g < 4; g++) {
            float s = red[0][lane][g * 8 + r];
#pragma unroll
            for (int w = 1; w < 4; w++) s += red[w][lane][g * 8 + r];
            gsum[g] = s + bvals[g];
        }
        const int grow = m0 + rbase + r;                  // batch index
        const float iv = sigmoid_f(gsum[0]);
        const float fv = sigmoid_f(gsum[1]);
        const float gv = tanh_f(gsum[2]);
        const float ov = sigmoid_f(gsum[3]);
        const int ci = grow * HH + gn;
        const float cnew = fv * c_st[ci] + iv * gv;
        const float hnew = ov * tanh_f(cnew);
        c_st[ci] = cnew;
        out[((size_t)grow * SS + t) * HH + gn] = hnew;    // outputs[b, t, :]
        h_out[ci] = f2bf_bits(hnew);
        if (t == SS - 1) {                                // final (h, c) tuple
            out[(size_t)BB * SS * HH + ci] = hnew;
            out[(size_t)BB * SS * HH + (size_t)BB * HH + ci] = cnew;
        }
    }
}

// ---------- host launch ----------
extern "C" void kernel_launch(void* const* d_in, const int* in_sizes, int n_in,
                              void* d_out, int out_size, void* d_ws, size_t ws_size,
                              hipStream_t stream) {
    const float* x  = (const float*)d_in[0];   // [B,S,I]
    const float* Wx = (const float*)d_in[1];   // [I,4H]
    const float* Wh = (const float*)d_in[2];   // [H,4H]
    const float* b  = (const float*)d_in[3];   // [4H]
    float* out = (float*)d_out;

    // workspace layout (all offsets 256B-aligned)
    char* ws = (char*)d_ws;
    unsigned short* x_bf = (unsigned short*)ws;                       // 33,554,432 B
    unsigned short* wx_p = (unsigned short*)(ws + 33554432);          //  4,194,304 B
    unsigned short* wh_p = (unsigned short*)(ws + 37748736);          //  8,388,608 B
    unsigned short* h_bf = (unsigned short*)(ws + 46137344);          //    262,144 B (2 buffers)
    float*          c_st = (float*)        (ws + 46399488);          //    262,144 B
    // total: 46,661,632 B (~44.5 MB) -- whole working set resident in 192 MB L2

    // 1) convert x to bf16 (4 elems/thread)
    {
        int n4 = (BB * SS * II) / 4;          // 4,194,304
        convert_x_kernel<<<(n4 + 255) / 256, 256, 0, stream>>>(x, x_bf, n4);
    }
    // 2) pre-pack weights into WMMA B-fragment order
    {
        int nk_x = II / 32, nk_h = HH / 32;
        int tx = 256 * nk_x * 32;             // 131,072 threads
        int th = 256 * nk_h * 32;             // 262,144 threads
        pack_w_kernel<<<(tx + 255) / 256, 256, 0, stream>>>(Wx, wx_p, nk_x);
        pack_w_kernel<<<(th + 255) / 256, 256, 0, stream>>>(Wh, wh_p, nk_h);
    }
    // 3) zero h (both buffers) and c
    init_state_kernel<<<(BB * HH + 255) / 256, 256, 0, stream>>>(h_bf, c_st);

    // 4) sequential scan: 512 step kernels, h double-buffered in bf16
    unsigned short* hA = h_bf;
    unsigned short* hB = h_bf + BB * HH;
    for (int t = 0; t < SS; t++) {
        const unsigned short* h_in = (t & 1) ? hB : hA;
        unsigned short* h_out      = (t & 1) ? hA : hB;
        lstm_step_kernel<<<256, 128, 0, stream>>>(x_bf, wx_p, wh_p, b,
                                                  h_in, h_out, c_st, out, t);
    }
}
